// EdgeAwareGNNEncoder_60610578481537
// MI455X (gfx1250) — compile-verified, matrix operand order checked by reading, not run on hardware
//
#include <hip/hip_runtime.h>
#include <hip/hip_bf16.h>
#include <math.h>

typedef __bf16 bf16_t;
typedef bf16_t v16bf __attribute__((ext_vector_type(16)));
typedef float  v8f   __attribute__((ext_vector_type(8)));

static __device__ __forceinline__ bf16_t f2bf(float f) { return (bf16_t)f; }

// ---------------- generic fill ----------------
__global__ void fill_kernel(float* p, float v, long n) {
  long i = (long)blockIdx.x * blockDim.x + threadIdx.x;
  if (i < n) p[i] = v;
}

// ---------------- input projection: out[n,:] = x[n,:K] @ w[K,128] + b ----------------
__global__ void proj_kernel(const float* __restrict__ x, const float* __restrict__ w,
                            const float* __restrict__ b, float* __restrict__ out,
                            int N, int K) {
  int idx = blockIdx.x * blockDim.x + threadIdx.x;
  if (idx >= N * 128) return;
  int n = idx >> 7, j = idx & 127;
  float acc = b[j];
  for (int k = 0; k < K; ++k) acc += x[n * K + k] * w[k * 128 + j];
  out[idx] = acc;
}

// ---------------- pack a [128,128] f32 weight into per-lane bf16 WMMA fragment order --------
// layout: [ntile(8)][kchunk(4)][lane(32)][elem(16)]
__global__ void pack_lin_kernel(const float* __restrict__ lin, bf16_t* __restrict__ out) {
  int idx = blockIdx.x * blockDim.x + threadIdx.x;
  if (idx >= 16384) return;
  int e     = idx & 15;
  int lane  = (idx >> 4) & 31;
  int kc    = (idx >> 9) & 3;
  int nt    = idx >> 11;
  int col   = nt * 16 + (lane & 15);
  int khalf = lane >> 4;
  int p = e >> 1, lo = e & 1;
  int k = kc * 32 + (p & 3) * 2 + khalf * 8 + ((p >> 2) << 4) + lo;
  out[idx] = f2bf(lin[k * 128 + col]);
}

// ---------------- WMMA bf16 GEMM, K=N=128: C[M,128] = A[M,128] @ B ----------------
// One wave per 16x16 tile; B pre-packed in fragment order (32B vector load / chunk).
__global__ __launch_bounds__(32)
void gemm_wmma128_kernel(const float* __restrict__ A, const bf16_t* __restrict__ Bp,
                         float* __restrict__ C, int M) {
  int lane  = threadIdx.x & 31;
  int r     = lane & 15;
  int khalf = lane >> 4;
  int m0 = blockIdx.x * 16;
  int nt = blockIdx.y;                       // 0..7
  int row = m0 + r; if (row >= M) row = M - 1;   // clamp (OOB rows never stored)
  const float* arow = A + (size_t)row * 128;
  v8f acc = {};
#pragma unroll
  for (int kc = 0; kc < 4; ++kc) {
    const float* ap = arow + kc * 32 + khalf * 8;
    float4 a0 = *(const float4*)(ap);
    float4 a1 = *(const float4*)(ap + 4);
    float4 a2 = *(const float4*)(ap + 16);
    float4 a3 = *(const float4*)(ap + 20);
    v16bf af;
    af[0]=f2bf(a0.x);  af[1]=f2bf(a0.y);  af[2]=f2bf(a0.z);  af[3]=f2bf(a0.w);
    af[4]=f2bf(a1.x);  af[5]=f2bf(a1.y);  af[6]=f2bf(a1.z);  af[7]=f2bf(a1.w);
    af[8]=f2bf(a2.x);  af[9]=f2bf(a2.y);  af[10]=f2bf(a2.z); af[11]=f2bf(a2.w);
    af[12]=f2bf(a3.x); af[13]=f2bf(a3.y); af[14]=f2bf(a3.z); af[15]=f2bf(a3.w);
    v16bf bfv = *(const v16bf*)(Bp + (((size_t)(nt * 4 + kc) * 32 + lane) << 4));
    acc = __builtin_amdgcn_wmma_f32_16x16x32_bf16(false, af, false, bfv,
                                                  (short)0, acc, false, false);
  }
  int col = nt * 16 + r;
  int mb  = m0 + khalf * 8;
  if (m0 + 16 <= M) {
#pragma unroll
    for (int t = 0; t < 8; ++t) C[(size_t)(mb + t) * 128 + col] = acc[t];
  } else {
#pragma unroll
    for (int t = 0; t < 8; ++t) { int m = mb + t; if (m < M) C[(size_t)m * 128 + col] = acc[t]; }
  }
}

// ---------------- per-head dot: out[m,h] = sum_c X[m, h*32+c] * att[h*32+c] ----------------
__global__ void headdot_rows_kernel(const float* __restrict__ X, const float* __restrict__ att,
                                    float* __restrict__ out, int M) {
  int idx = blockIdx.x * blockDim.x + threadIdx.x;
  if (idx >= M * 4) return;
  int m = idx >> 2, h = idx & 3;
  float s = 0.f;
  for (int c = 0; c < 32; ++c) s += X[(size_t)m * 128 + h * 32 + c] * att[h * 32 + c];
  out[idx] = s;
}

// ---------------- out[q,h] = sum_j A[q*128+j] * W[j*4+h] ----------------
__global__ void gemm4_kernel(const float* __restrict__ A, const float* __restrict__ W,
                             float* __restrict__ out, int Q) {
  int idx = blockIdx.x * blockDim.x + threadIdx.x;
  if (idx >= Q * 4) return;
  int q = idx >> 2, h = idx & 3;
  float s = 0.f;
  for (int j = 0; j < 128; ++j) s += A[(size_t)q * 128 + j] * W[j * 4 + h];
  out[idx] = s;
}

__global__ void addvec_kernel(float* __restrict__ p, const float* __restrict__ q, int n) {
  int i = blockIdx.x * blockDim.x + threadIdx.x;
  if (i < n) p[i] += q[i];
}

static __device__ __forceinline__ void atomicMaxF(float* addr, float v) {
  if (v >= 0.f) atomicMax((int*)addr, __float_as_int(v));
  else          atomicMin((unsigned int*)addr, __float_as_uint(v));
}

// ---------------- edge attention logits + segment max ----------------
__global__ void edge_a_kernel(const float* __restrict__ ea, const int* __restrict__ src,
                              const int* __restrict__ dst,
                              const float* __restrict__ as_, const float* __restrict__ ad_,
                              const float* __restrict__ w1, const float* __restrict__ b1,
                              const float* __restrict__ w2ae, const float* __restrict__ b2ae,
                              float* __restrict__ aout, float* __restrict__ amax, int E) {
  int e = blockIdx.x * blockDim.x + threadIdx.x;
  if (e >= E) return;
  float e0 = ea[e * 3], e1 = ea[e * 3 + 1], e2 = ea[e * 3 + 2];
  float acc[4] = {b2ae[0], b2ae[1], b2ae[2], b2ae[3]};
  for (int q = 0; q < 64; ++q) {
    float hm = e0 * w1[q] + e1 * w1[64 + q] + e2 * w1[128 + q] + b1[q];
    hm = hm > 0.f ? hm : 0.f;
    acc[0] += hm * w2ae[q * 4 + 0];
    acc[1] += hm * w2ae[q * 4 + 1];
    acc[2] += hm * w2ae[q * 4 + 2];
    acc[3] += hm * w2ae[q * 4 + 3];
  }
  int s = src[e], d = dst[e];
#pragma unroll
  for (int h = 0; h < 4; ++h) {
    float a = as_[s * 4 + h] + ad_[d * 4 + h] + acc[h];
    a = a > 0.f ? a : 0.2f * a;               // leaky_relu(0.2)
    aout[e * 4 + h] = a;
    atomicMaxF(&amax[d * 4 + h], a);
  }
}

// ---------------- exp(a - amax[dst]) + segment sum ----------------
__global__ void edge_exp_kernel(const float* __restrict__ aout, const int* __restrict__ dst,
                                const float* __restrict__ amax, float* __restrict__ ex,
                                float* __restrict__ den, int E) {
  int idx = blockIdx.x * blockDim.x + threadIdx.x;
  if (idx >= E * 4) return;
  int e = idx >> 2, h = idx & 3;
  int d = dst[e];
  float v = expf(aout[idx] - amax[d * 4 + h]);
  ex[idx] = v;
  atomicAdd(&den[d * 4 + h], v);
}

// ---------------- scatter messages: agg[dst] += xs[src] * alpha ----------------
__global__ void edge_msg_kernel(const float* __restrict__ ex, const float* __restrict__ den,
                                const int* __restrict__ src, const int* __restrict__ dst,
                                const float* __restrict__ xs, float* __restrict__ agg, int E) {
  long tid = (long)blockIdx.x * blockDim.x + threadIdx.x;
  if (tid >= (long)E * 32) return;
  int e = (int)(tid >> 5);
  int lane = (int)(tid & 31);
  int s = src[e], d = dst[e];
#pragma unroll
  for (int h = 0; h < 4; ++h) {
    int ch = lane + h * 32;                   // channel ch belongs to head h
    float alpha = ex[e * 4 + h] / (den[d * 4 + h] + 1e-16f);
    atomicAdd(&agg[(size_t)d * 128 + ch], xs[(size_t)s * 128 + ch] * alpha);
  }
}

// ---------------- combine: LN(relu((agg+biasSum)/R) + h) ----------------
__global__ __launch_bounds__(32)
void combine_kernel(const float* __restrict__ h, const float* __restrict__ agg,
                    const float* __restrict__ bsum, float invR,
                    const float* __restrict__ g, const float* __restrict__ bb,
                    float* __restrict__ out, int N) {
  int n = blockIdx.x;
  if (n >= N) return;
  int lane = threadIdx.x;
  float vals[4];
  float s = 0.f;
#pragma unroll
  for (int t = 0; t < 4; ++t) {
    int ch = lane + t * 32;
    float x = (agg[(size_t)n * 128 + ch] + bsum[ch]) * invR;
    x = x > 0.f ? x : 0.f;
    x += h[(size_t)n * 128 + ch];
    vals[t] = x;
    s += x;
  }
  for (int off = 16; off; off >>= 1) s += __shfl_xor(s, off, 32);
  float mean = s * (1.0f / 128.0f);
  float v = 0.f;
#pragma unroll
  for (int t = 0; t < 4; ++t) { float d = vals[t] - mean; v += d * d; }
  for (int off = 16; off; off >>= 1) v += __shfl_xor(v, off, 32);
  float inv = rsqrtf(v * (1.0f / 128.0f) + 1e-5f);
#pragma unroll
  for (int t = 0; t < 4; ++t) {
    int ch = lane + t * 32;
    out[(size_t)n * 128 + ch] = (vals[t] - mean) * inv * g[ch] + bb[ch];
  }
}

// ---------------- pooling ----------------
__global__ void pool_cnt_kernel(const int* __restrict__ batch, float* __restrict__ cnt, int N) {
  int n = blockIdx.x * blockDim.x + threadIdx.x;
  if (n < N) atomicAdd(&cnt[batch[n]], 1.0f);
}
__global__ void pool_sum_kernel(const float* __restrict__ h, const int* __restrict__ batch,
                                float* __restrict__ psum, int coff, int N) {
  int idx = blockIdx.x * blockDim.x + threadIdx.x;
  if (idx >= N * 128) return;
  int n = idx >> 7, ch = idx & 127;
  int b = batch[n];
  atomicAdd(&psum[(size_t)b * 512 + coff + ch], h[idx]);
}
__global__ void pool_div_kernel(const float* __restrict__ psum, const float* __restrict__ cnt,
                                float* __restrict__ out) {
  int idx = blockIdx.x * blockDim.x + threadIdx.x;
  if (idx >= 64 * 512) return;
  int b = idx >> 9;
  int t = (idx & 511) >> 7;
  out[idx] = psum[idx] / fmaxf(cnt[t * 64 + b], 1.0f);
}

// =======================================================================
static inline int cdiv(long a, long b) { return (int)((a + b - 1) / b); }

extern "C" void kernel_launch(void* const* d_in, const int* in_sizes, int n_in,
                              void* d_out, int out_size, void* d_ws, size_t ws_size,
                              hipStream_t stream) {
  (void)n_in; (void)out_size; (void)ws_size;
  const int NT = 4;
  const int rs[8] = {0, 1, 0, 2, 0, 0, 1, 2};
  const int rd[8] = {1, 0, 2, 0, 3, 0, 3, 3};
  const int IN_DIM[4] = {8, 10, 7, 3};
  const int nrel_dst[4] = {3, 1, 1, 3};

  const float* x_t[4];  const int* b_t[4];
  int Ncnt[4];
  for (int t = 0; t < NT; ++t) {
    x_t[t] = (const float*)d_in[2 * t];
    b_t[t] = (const int*)d_in[2 * t + 1];
    Ncnt[t] = in_sizes[2 * t + 1];
  }
  const int* ei[8]; const float* eaP[8]; int Ecnt[8];
  for (int i = 0; i < 8; ++i) {
    ei[i]  = (const int*)d_in[8 + 2 * i];
    eaP[i] = (const float*)d_in[9 + 2 * i];
    Ecnt[i] = in_sizes[9 + 2 * i] / 3;
  }
  // params (insertion-order flattening)
  const float* proj_w[4]; const float* proj_b[4];
  for (int t = 0; t < NT; ++t) { proj_w[t] = (const float*)d_in[24 + 2 * t]; proj_b[t] = (const float*)d_in[25 + 2 * t]; }
  const float* ee1w = (const float*)d_in[32];
  const float* ee1b = (const float*)d_in[33];
  const float* ee2w = (const float*)d_in[34];
  const float* ee2b = (const float*)d_in[35];
  auto relbase = [](int l, int i) { return 36 + l * 50 + i * 6; };

  // node offsets in concatenated h
  int hoff[4]; { int o = 0; for (int t = 0; t < NT; ++t) { hoff[t] = o; o += Ncnt[t]; } }
  long NTOT = hoff[3] + Ncnt[3];

  // ---- workspace carve (floats) ----
  float* W = (float*)d_ws;
  size_t o = 0;
  float* hA   = W + o; o += (size_t)NTOT * 128;
  float* hB   = W + o; o += (size_t)NTOT * 128;
  float* agg  = W + o; o += (size_t)NTOT * 128;
  float* xs   = W + o; o += (size_t)50000 * 128;
  float* as_  = W + o; o += 50000 * 4;
  float* ad_  = W + o; o += 50000 * 4;
  float* amax = W + o; o += 50000 * 4;
  float* den  = W + o; o += 50000 * 4;
  float* abuf = W + o; o += 200000 * 4;
  float* exbf = W + o; o += 200000 * 4;
  float* w2ae = W + o; o += 16 * 64 * 4;
  float* b2ae = W + o; o += 16 * 4;
  float* wtmp = W + o; o += 128 * 4;
  float* wad  = W + o; o += 128 * 4;
  float* bsum = W + o; o += 4 * 128;
  float* psum = W + o; o += 64 * 512;
  float* pcnt = W + o; o += 4 * 64;
  bf16_t* linpk = (bf16_t*)(W + o); o += 16 * 16384 / 2;   // 16 weights, bf16 fragment-packed

  const int TB = 256;
  const float NEG_INF = -__builtin_inff();

  // 1. projections
  for (int t = 0; t < NT; ++t)
    proj_kernel<<<cdiv((long)Ncnt[t] * 128, TB), TB, 0, stream>>>(
        x_t[t], proj_w[t], proj_b[t], hA + (size_t)hoff[t] * 128, Ncnt[t], IN_DIM[t]);

  // 2. precompute folded edge-attention weights per (layer, relation) and pack lin -> bf16
  for (int li = 0; li < 16; ++li) {
    int l = li >> 3, i = li & 7, rb = relbase(l, i);
    const float* lin      = (const float*)d_in[rb + 0];
    const float* lin_edge = (const float*)d_in[rb + 1];
    const float* att_edge = (const float*)d_in[rb + 4];
    pack_lin_kernel<<<cdiv(16384, TB), TB, 0, stream>>>(lin, linpk + (size_t)li * 16384);
    headdot_rows_kernel<<<cdiv(128 * 4, TB), TB, 0, stream>>>(lin_edge, att_edge, wtmp, 128);
    gemm4_kernel<<<cdiv(64 * 4, TB), TB, 0, stream>>>(ee2w, wtmp, w2ae + li * 256, 64);
    gemm4_kernel<<<1, TB, 0, stream>>>(ee2b, wtmp, b2ae + li * 4, 1);
  }

  // 3. GNN layers
  for (int l = 0; l < 2; ++l) {
    float* hcur = l ? hB : hA;
    float* hnxt = l ? hA : hB;
    fill_kernel<<<cdiv(NTOT * 128, TB), TB, 0, stream>>>(agg, 0.f, NTOT * 128);

    for (int i = 0; i < 8; ++i) {
      int rb = relbase(l, i), s = rs[i], d = rd[i];
      int Ns = Ncnt[s], Nd = Ncnt[d], E = Ecnt[i];
      const float* lin     = (const float*)d_in[rb + 0];
      const float* att_src = (const float*)d_in[rb + 2];
      const float* att_dst = (const float*)d_in[rb + 3];
      const int* src = ei[i];
      const int* dst = ei[i] + E;
      int li = l * 8 + i;

      // xs = h[s] @ lin  (WMMA bf16, f32 accumulate, pre-packed B)
      gemm_wmma128_kernel<<<dim3(cdiv(Ns, 16), 8), 32, 0, stream>>>(
          hcur + (size_t)hoff[s] * 128, linpk + (size_t)li * 16384, xs, Ns);
      // a_src[n,h] = sum_c xs[n,h,c]*att_src[h,c]
      headdot_rows_kernel<<<cdiv((long)Ns * 4, TB), TB, 0, stream>>>(xs, att_src, as_, Ns);
      // Wad[128,4] = lin reduced by att_dst ; a_dst = h[d] @ Wad
      headdot_rows_kernel<<<cdiv(128 * 4, TB), TB, 0, stream>>>(lin, att_dst, wad, 128);
      gemm4_kernel<<<cdiv((long)Nd * 4, TB), TB, 0, stream>>>(hcur + (size_t)hoff[d] * 128, wad, ad_, Nd);

      fill_kernel<<<cdiv((long)Nd * 4, TB), TB, 0, stream>>>(amax, NEG_INF, (long)Nd * 4);
      fill_kernel<<<cdiv((long)Nd * 4, TB), TB, 0, stream>>>(den, 0.f, (long)Nd * 4);

      edge_a_kernel<<<cdiv(E, TB), TB, 0, stream>>>(
          eaP[i], src, dst, as_, ad_, ee1w, ee1b, w2ae + li * 256, b2ae + li * 4, abuf, amax, E);
      edge_exp_kernel<<<cdiv((long)E * 4, TB), TB, 0, stream>>>(abuf, dst, amax, exbf, den, E);
      edge_msg_kernel<<<cdiv((long)E * 32, TB), TB, 0, stream>>>(
          exbf, den, src, dst, xs, agg + (size_t)hoff[d] * 128, E);
    }

    // per-type: sum relation biases, then combine + layernorm
    const float* g  = (const float*)d_in[36 + l * 50 + 48];
    const float* bb = (const float*)d_in[36 + l * 50 + 49];
    for (int t = 0; t < NT; ++t) {
      fill_kernel<<<1, 128, 0, stream>>>(bsum + t * 128, 0.f, 128);
      for (int i = 0; i < 8; ++i)
        if (rd[i] == t)
          addvec_kernel<<<1, 128, 0, stream>>>(bsum + t * 128, (const float*)d_in[relbase(l, i) + 5], 128);
      combine_kernel<<<Ncnt[t], 32, 0, stream>>>(
          hcur + (size_t)hoff[t] * 128, agg + (size_t)hoff[t] * 128, bsum + t * 128,
          1.0f / (float)nrel_dst[t], g, bb, hnxt + (size_t)hoff[t] * 128, Ncnt[t]);
    }
  }

  // 4. global mean pool per type (final h is in hA after 2 layers)
  fill_kernel<<<cdiv(64 * 512, TB), TB, 0, stream>>>(psum, 0.f, 64 * 512);
  fill_kernel<<<1, 256, 0, stream>>>(pcnt, 0.f, 256);
  for (int t = 0; t < NT; ++t) {
    pool_cnt_kernel<<<cdiv(Ncnt[t], TB), TB, 0, stream>>>(b_t[t], pcnt + t * 64, Ncnt[t]);
    pool_sum_kernel<<<cdiv((long)Ncnt[t] * 128, TB), TB, 0, stream>>>(
        hA + (size_t)hoff[t] * 128, b_t[t], psum, t * 128, Ncnt[t]);
  }
  pool_div_kernel<<<cdiv(64 * 512, TB), TB, 0, stream>>>(psum, pcnt, (float*)d_out);
}
